// RoPEAttention_70153995812979
// MI455X (gfx1250) — matrix operand order repaired and checked by reading, not compiled, not run on hardware
//
#include <hip/hip_runtime.h>
#include <hip/hip_bf16.h>
#include <stdint.h>

// Problem constants (match the reference).
#define B_   2
#define S_   2048
#define HID_ 2048
#define H_   16
#define KV_  4
#define D_   128

typedef __attribute__((ext_vector_type(16))) __bf16 v16bf;
typedef __attribute__((ext_vector_type(8)))  float  v8f;
typedef __attribute__((ext_vector_type(4)))  float  v4f;

union Frag16 {            // 16 bf16 = one WMMA A/B operand (8 VGPRs)
  v16bf          v;
  v4f            f[2];    // two 16-byte chunks
  unsigned short u[16];
};

__device__ __forceinline__ unsigned short f2bf(float x) {
  unsigned int u = __builtin_bit_cast(unsigned int, x);
  u += 0x7fffu + ((u >> 16) & 1u);          // round-to-nearest-even
  return (unsigned short)(u >> 16);
}

__device__ __forceinline__ v8f v8f_zero() {
  v8f z;
#pragma unroll
  for (int i = 0; i < 8; ++i) z[i] = 0.0f;
  return z;
}

__device__ __forceinline__ v8f wmma_bf16(v16bf a, v16bf b, v8f c) {
  // D = A(16x32 bf16) * B(32x16 bf16) + C(16x16 f32)
  return __builtin_amdgcn_wmma_f32_16x16x32_bf16(false, a, false, b, (short)0, c,
                                                 false, false);
}

// Async 16B global -> LDS copy (tracked by ASYNCcnt; in-order completion).
// ldsOff = low 32 bits of the shared-memory flat address (LDS byte offset).
__device__ __forceinline__ void async_copy_b128(unsigned ldsOff,
                                                const void* gaddr) {
  asm volatile("global_load_async_to_lds_b128 %0, %1, off"
               :: "v"(ldsOff), "v"((unsigned long long)(uintptr_t)gaddr)
               : "memory");
}
// Wait until ASYNCcnt <= n (immediate).
#define WAIT_ASYNC(n) asm volatile("s_wait_asynccnt " #n ::: "memory")

__device__ __forceinline__ unsigned lds_off(const void* p) {
  return (unsigned)(uintptr_t)p;  // low 32 bits of flat shared addr = LDS offset
}

// ---------------------------------------------------------------------------
// 1a) fp32 -> bf16 elementwise conversion
// ---------------------------------------------------------------------------
__global__ __launch_bounds__(256) void cvt_bf16_kernel(
    const float* __restrict__ src, unsigned short* __restrict__ dst, int n) {
  int i = blockIdx.x * 256 + threadIdx.x;
  if (i < n) dst[i] = f2bf(src[i]);
}

// ---------------------------------------------------------------------------
// 1b) fp32 -> bf16 transposing conversion: src [K,N] -> dst [N,K].
//     Weights are transposed ONCE here so every GEMM B-panel becomes a
//     contiguous async copy (no per-panel LDS transpose in the hot loop).
// ---------------------------------------------------------------------------
__global__ __launch_bounds__(256) void cvt_bf16_t_kernel(
    const float* __restrict__ src, unsigned short* __restrict__ dst,
    int K, int N) {
  int idx = blockIdx.x * 256 + threadIdx.x;
  if (idx >= K * N) return;
  const int n = idx % N;
  const int k = idx / N;
  dst[(size_t)n * K + k] = f2bf(src[(size_t)k * N + n]);
}

// ---------------------------------------------------------------------------
// 2) Tiled bf16 GEMM: C(f32)[M,N] = A(bf16)[M,K] @ Bt(bf16)[N,K]^T
//    256 threads = 8 waves; block tile 128x64; wave tile 32x32 (2x2 WMMA).
//    K-panel of 64, DOUBLE-BUFFERED: panel k+1 is prefetched with
//    global_load_async_to_lds_b128 while panel k is consumed; each thread
//    issues exactly 3 async chunks/panel, so "s_wait_asynccnt 3" retires the
//    previous buffer (in-order completion) without draining the prefetch.
//    Requires 128|M, 64|N, 64|K (true for all our shapes).
// ---------------------------------------------------------------------------
__global__ __launch_bounds__(256) void gemm_bf16_f32_kernel(
    const unsigned short* __restrict__ A,   // [M,K] row-major
    const unsigned short* __restrict__ Bt,  // [N,K] row-major (pre-transposed)
    float* __restrict__ C, int M, int N, int K) {
  __shared__ unsigned short As[2][128][72];  // 72 = 64 + pad (16B-aligned rows)
  __shared__ unsigned short Bs[2][64][72];   // stored [n][k]

  const int tid  = threadIdx.x;
  const int lane = tid & 31;
  const int wave = tid >> 5;
  const int m15  = lane & 15;
  const int hlf  = lane >> 4;             // half-wave select
  const int bM   = blockIdx.y * 128;
  const int bN   = blockIdx.x * 64;
  const int wr   = wave >> 1;             // 0..3  (row tile of 32)
  const int wc   = wave & 1;              // 0..1  (col tile of 32)

  v8f acc[2][2];
#pragma unroll
  for (int i = 0; i < 2; ++i)
#pragma unroll
    for (int j = 0; j < 2; ++j) acc[i][j] = v8f_zero();

  // Staging: 512 A-chunks (128 rows x 4) + 256 B-chunks (64 rows x 4)
  // over 256 threads -> exactly 3 x 16B async copies per thread per panel.
  const int srow = tid >> 2;              // 0..63
  const int scol = (tid & 3) * 8;         // 0,8,16,24 (bf16 elems)
  unsigned aL0[2], aL1[2], bL[2];
#pragma unroll
  for (int i = 0; i < 2; ++i) {
    aL0[i] = lds_off(&As[i][srow][scol]);
    aL1[i] = lds_off(&As[i][srow + 64][scol]);
    bL[i]  = lds_off(&Bs[i][srow][scol]);
  }
  const unsigned short* aG0 = A  + (size_t)(bM + srow) * K + scol;
  const unsigned short* aG1 = A  + (size_t)(bM + srow + 64) * K + scol;
  const unsigned short* bG  = Bt + (size_t)(bN + srow) * K + scol;

  // Prologue: fill buffer 0.
  async_copy_b128(aL0[0], aG0);
  async_copy_b128(aL1[0], aG1);
  async_copy_b128(bL[0], bG);

  int buf = 0;
  for (int kb = 0; kb < K; kb += 64) {
    const int nxt = kb + 64;
    if (nxt < K) {  // prefetch next panel into the other buffer
      async_copy_b128(aL0[buf ^ 1], aG0 + nxt);
      async_copy_b128(aL1[buf ^ 1], aG1 + nxt);
      async_copy_b128(bL[buf ^ 1], bG + nxt);
      WAIT_ASYNC(3);      // current buffer's 3 copies retired; prefetch flies
    } else {
      WAIT_ASYNC(0);
    }
    __syncthreads();

#pragma unroll
    for (int ks = 0; ks < 2; ++ks) {      // two 32-wide K-steps per panel
      Frag16 af[2], bf[2];
#pragma unroll
      for (int t = 0; t < 2; ++t) {
        // A fragment: row m = lane&15; e<8 -> k=8h+e, e>=8 -> k=16+8h+(e-8)
        const int r = wr * 32 + t * 16 + m15;
        af[t].f[0] = *(const v4f*)&As[buf][r][ks * 32 + 8 * hlf];
        af[t].f[1] = *(const v4f*)&As[buf][r][ks * 32 + 16 + 8 * hlf];
        // B fragment: col n = lane&15; e<8 -> k=8h+e, e>=8 -> k=16+8h+(e-8)
        const int c = wc * 32 + t * 16 + m15;
        bf[t].f[0] = *(const v4f*)&Bs[buf][c][ks * 32 + 8 * hlf];
        bf[t].f[1] = *(const v4f*)&Bs[buf][c][ks * 32 + 16 + 8 * hlf];
      }
#pragma unroll
      for (int i = 0; i < 2; ++i)
#pragma unroll
        for (int j = 0; j < 2; ++j)
          acc[i][j] = wmma_bf16(af[i].v, bf[j].v, acc[i][j]);
    }
    __syncthreads();   // reads done before this buffer is overwritten
    buf ^= 1;
  }

  // Epilogue: C layout -> lane holds col N=lane&15, rows M = r + 8*half
#pragma unroll
  for (int i = 0; i < 2; ++i)
#pragma unroll
    for (int j = 0; j < 2; ++j) {
      const int col = bN + wc * 32 + j * 16 + m15;
#pragma unroll
      for (int r = 0; r < 8; ++r) {
        const int row = bM + wr * 32 + i * 16 + r + 8 * hlf;
        C[(size_t)row * N + col] = acc[i][j][r];
      }
    }
}

// ---------------------------------------------------------------------------
// 3) RoPE + head transpose: src f32 [B*S, nh*D] -> dst bf16 [B, nh, S, D]
//    out[j]    = x[2j]*cos - x[2j+1]*sin
//    out[64+j] = x[2j]*sin + x[2j+1]*cos          (j = 0..63)
// ---------------------------------------------------------------------------
__global__ __launch_bounds__(256) void rope_kernel(
    const float* __restrict__ src, unsigned short* __restrict__ dst,
    int nh, int n) {
  int idx = blockIdx.x * 256 + threadIdx.x;
  if (idx >= n) return;
  const int j  = idx & 63;
  int rest     = idx >> 6;
  const int hh = rest % nh;  rest /= nh;
  const int s  = rest % S_;
  const int b  = rest / S_;
  const float* row = src + ((size_t)(b * S_ + s)) * (nh * D_) + hh * D_;
  const float x1 = row[2 * j], x2 = row[2 * j + 1];
  const float inv = __powf(10000.0f, -(float)(2 * j) * (1.0f / 128.0f));
  float sn, cs;
  __sincosf((float)s * inv, &sn, &cs);
  unsigned short* orow = dst + (((size_t)(b * nh + hh)) * S_ + s) * D_;
  orow[j]      = f2bf(x1 * cs - x2 * sn);
  orow[64 + j] = f2bf(x1 * sn + x2 * cs);
}

// ---------------------------------------------------------------------------
// 4) V transpose: src f32 [B*S, KV*D] -> dst bf16 [B, KV, D, S]
// ---------------------------------------------------------------------------
__global__ __launch_bounds__(256) void vtrans_kernel(
    const float* __restrict__ src, unsigned short* __restrict__ dst, int n) {
  int idx = blockIdx.x * 256 + threadIdx.x;
  if (idx >= n) return;
  const int d  = idx & (D_ - 1);
  int rest     = idx >> 7;
  const int hh = rest % KV_;  rest /= KV_;
  const int s  = rest % S_;
  const int b  = rest / S_;
  const float v = src[((size_t)(b * S_ + s)) * (KV_ * D_) + hh * D_ + d];
  dst[(((size_t)(b * KV_ + hh)) * D_ + d) * S_ + s] = f2bf(v);
}

// ---------------------------------------------------------------------------
// 5) Causal GQA flash attention (wave32, one wave per 16-row Q tile).
//    S^T = K @ Q^T  (so each lane owns one fixed Q column for softmax);
//    its accumulator layout (kv = r + 8*half) IS the A-fragment layout of P
//    (kv = 8h+e / 16+8h+e), so exp() feeds the PV WMMA with no shuffles.
// ---------------------------------------------------------------------------
__global__ __launch_bounds__(128) void flash_attn_kernel(
    const unsigned short* __restrict__ qT,  // [B,H,S,D]  (RoPE'd)
    const unsigned short* __restrict__ kT,  // [B,KV,S,D] (RoPE'd)
    const unsigned short* __restrict__ vT,  // [B,KV,D,S]
    unsigned short* __restrict__ O) {       // [B*S, H*D]
  const int lane = threadIdx.x & 31;
  const int wave = threadIdx.x >> 5;
  const int m15  = lane & 15;
  const int hlf  = lane >> 4;

  const int qg    = blockIdx.x & 31;       // 32 q-groups of 64 rows
  const int bh    = blockIdx.x >> 5;       // B*H = 32 combos
  const int head  = bh & (H_ - 1);
  const int b     = bh >> 4;
  const int kvh   = head >> 2;             // GQA: H/KV = 4
  const int qbase = qg * 64 + wave * 16;

  const unsigned short* qp = qT + ((size_t)(b * H_ + head) * S_) * D_;
  const unsigned short* kp = kT + ((size_t)(b * KV_ + kvh) * S_) * D_;
  const unsigned short* vp = vT + ((size_t)(b * KV_ + kvh) * D_) * S_;

  // Q fragments (B operand of S^T): col q = lane&15, elements d = dc*32+{8h+e, 16+8h+(e-8)}
  Frag16 qf[4];
  const int qrow = qbase + m15;
#pragma unroll
  for (int dc = 0; dc < 4; ++dc) {
    const unsigned short* p = qp + (size_t)qrow * D_ + dc * 32 + 8 * hlf;
    qf[dc].f[0] = *(const v4f*)p;
    qf[dc].f[1] = *(const v4f*)(p + 16);
  }

  v8f oacc[8];
#pragma unroll
  for (int i = 0; i < 8; ++i) oacc[i] = v8f_zero();
  float mrun = -INFINITY, lrun = 0.0f;
  const float scale = 0.08838834764831845f;  // 1/sqrt(128)
  const int qpos = qbase + m15;              // this lane's Q column

  const int nblocks = (qbase >> 5) + 1;      // causal KV extent, 32 per block
  for (int kb = 0; kb < nblocks; ++kb) {
    const int kvbase = kb * 32;
    // --- S^T = K(16x32d per tile) @ Q^T ---
    v8f s0 = v8f_zero(), s1 = v8f_zero();
#pragma unroll
    for (int dc = 0; dc < 4; ++dc) {
      Frag16 k0, k1;  // A operand: row kv = lane&15, d = dc*32 + {8h..,16+8h..}
      const unsigned short* p0 = kp + (size_t)(kvbase + m15) * D_ + dc * 32;
      const unsigned short* p1 = kp + (size_t)(kvbase + 16 + m15) * D_ + dc * 32;
      k0.f[0] = *(const v4f*)(p0 + 8 * hlf);
      k0.f[1] = *(const v4f*)(p0 + 16 + 8 * hlf);
      k1.f[0] = *(const v4f*)(p1 + 8 * hlf);
      k1.f[1] = *(const v4f*)(p1 + 16 + 8 * hlf);
      s0 = wmma_bf16(k0.v, qf[dc].v, s0);
      s1 = wmma_bf16(k1.v, qf[dc].v, s1);
    }
    // --- scale + causal mask (acc element r holds kv = kvbase + 8h + r) ---
    float tmax = -INFINITY;
#pragma unroll
    for (int r = 0; r < 8; ++r) {
      const int kv0 = kvbase + 8 * hlf + r;
      s0[r] = (kv0 <= qpos) ? s0[r] * scale : -INFINITY;
      s1[r] = (kv0 + 16 <= qpos) ? s1[r] * scale : -INFINITY;
      tmax = fmaxf(tmax, fmaxf(s0[r], s1[r]));
    }
    tmax = fmaxf(tmax, __shfl_xor(tmax, 16, 32));
    const float mnew   = fmaxf(mrun, tmax);
    const float factor = __expf(mrun - mnew);
    // --- P = exp(S - m), packed directly as next A fragment ---
    Frag16 pf;
    float rsum = 0.0f;
#pragma unroll
    for (int r = 0; r < 8; ++r) {
      const float p0 = __expf(s0[r] - mnew);
      const float p1 = __expf(s1[r] - mnew);
      rsum += p0 + p1;
      pf.u[r]     = f2bf(p0);
      pf.u[8 + r] = f2bf(p1);
    }
    rsum += __shfl_xor(rsum, 16, 32);
    lrun = lrun * factor + rsum;
    mrun = mnew;
    // --- rescale O acc (rows M = r + 8h need factor of q = r + 8h) ---
    float fr[8];
#pragma unroll
    for (int r = 0; r < 8; ++r) fr[r] = __shfl(factor, 8 * hlf + r, 32);
#pragma unroll
    for (int dc = 0; dc < 8; ++dc)
#pragma unroll
      for (int r = 0; r < 8; ++r) oacc[dc][r] *= fr[r];
    // --- O += P @ V : B operand from pre-transposed V (contiguous in s) ---
#pragma unroll
    for (int dc = 0; dc < 8; ++dc) {
      Frag16 vfr;  // col d = dc*16 + lane&15, elements kv = 8h+e / 16+8h+(e-8)
      const unsigned short* p =
          vp + (size_t)(dc * 16 + m15) * S_ + kvbase + 8 * hlf;
      vfr.f[0] = *(const v4f*)p;
      vfr.f[1] = *(const v4f*)(p + 16);
      oacc[dc] = wmma_bf16(pf.v, vfr.v, oacc[dc]);
    }
  }

  // Normalize by l and store: O layout lane -> col d, rows q = r + 8h
  float lr[8];
#pragma unroll
  for (int r = 0; r < 8; ++r) lr[r] = __shfl(lrun, 8 * hlf + r, 32);
#pragma unroll
  for (int dc = 0; dc < 8; ++dc) {
    const int d = dc * 16 + m15;
#pragma unroll
    for (int r = 0; r < 8; ++r) {
      const int q = qbase + r + 8 * hlf;
      O[((size_t)(b * S_ + q)) * (H_ * D_) + head * D_ + d] =
          f2bf(oacc[dc][r] / lr[r]);
    }
  }
}

// ---------------------------------------------------------------------------
// Launch: cvt (weights transposed) -> QKV GEMMs -> RoPE/transposes ->
//         flash attention -> Wo GEMM.
// Workspace layout (bytes, all 1MB-aligned; total 124 MB):
//   xb 0..16M | WqT 16..24 | WkT 24..26 | WvT 26..28 | WoT 28..36 |
//   qf 36..68 | kf 68..76 | vf 76..84 | qT 84..100 | kT 100..104 |
//   vT 104..108 | ao 108..124
// ---------------------------------------------------------------------------
extern "C" void kernel_launch(void* const* d_in, const int* in_sizes, int n_in,
                              void* d_out, int out_size, void* d_ws,
                              size_t ws_size, hipStream_t stream) {
  (void)in_sizes; (void)n_in; (void)out_size; (void)ws_size;
  const float* x  = (const float*)d_in[0];
  /* d_in[1] = mask (unused: causal structure is known) */
  const float* Wq = (const float*)d_in[2];
  const float* Wk = (const float*)d_in[3];
  const float* Wv = (const float*)d_in[4];
  const float* Wo = (const float*)d_in[5];
  float* out = (float*)d_out;

  char* ws = (char*)d_ws;
  const size_t MB = 1ull << 20;
  unsigned short* xb  = (unsigned short*)(ws + 0 * MB);
  unsigned short* wqT = (unsigned short*)(ws + 16 * MB);
  unsigned short* wkT = (unsigned short*)(ws + 24 * MB);
  unsigned short* wvT = (unsigned short*)(ws + 26 * MB);
  unsigned short* woT = (unsigned short*)(ws + 28 * MB);
  float*          qf  = (float*)(ws + 36 * MB);
  float*          kf  = (float*)(ws + 68 * MB);
  float*          vf  = (float*)(ws + 76 * MB);
  unsigned short* qT  = (unsigned short*)(ws + 84 * MB);
  unsigned short* kT  = (unsigned short*)(ws + 100 * MB);
  unsigned short* vT  = (unsigned short*)(ws + 104 * MB);
  unsigned short* ao  = (unsigned short*)(ws + 108 * MB);

  const int M = B_ * S_;  // 4096

  // 1) fp32 -> bf16 (x straight; weights transposed to [N][K])
  {
    int n = M * HID_;
    cvt_bf16_kernel<<<(n + 255) / 256, 256, 0, stream>>>(x, xb, n);
  }
  auto cvtT = [&](const float* s, unsigned short* d, int K, int N) {
    int n = K * N;
    cvt_bf16_t_kernel<<<(n + 255) / 256, 256, 0, stream>>>(s, d, K, N);
  };
  cvtT(Wq, wqT, HID_, H_ * D_);
  cvtT(Wk, wkT, HID_, KV_ * D_);
  cvtT(Wv, wvT, HID_, KV_ * D_);
  cvtT(Wo, woT, H_ * D_, HID_);

  // 2) QKV projections (bf16 WMMA, double-buffered async-LDS staging)
  gemm_bf16_f32_kernel<<<dim3((H_ * D_) / 64, M / 128), 256, 0, stream>>>(
      xb, wqT, qf, M, H_ * D_, HID_);
  gemm_bf16_f32_kernel<<<dim3((KV_ * D_) / 64, M / 128), 256, 0, stream>>>(
      xb, wkT, kf, M, KV_ * D_, HID_);
  gemm_bf16_f32_kernel<<<dim3((KV_ * D_) / 64, M / 128), 256, 0, stream>>>(
      xb, wvT, vf, M, KV_ * D_, HID_);

  // 3) RoPE (q,k) + V transpose, all emitting bf16 in attention layouts
  {
    int nq = B_ * S_ * H_ * 64;
    rope_kernel<<<(nq + 255) / 256, 256, 0, stream>>>(qf, qT, H_, nq);
    int nk = B_ * S_ * KV_ * 64;
    rope_kernel<<<(nk + 255) / 256, 256, 0, stream>>>(kf, kT, KV_, nk);
    int nv = B_ * S_ * KV_ * D_;
    vtrans_kernel<<<(nv + 255) / 256, 256, 0, stream>>>(vf, vT, nv);
  }

  // 4) Causal GQA flash attention: 1024 blocks x 128 threads (4 waves)
  flash_attn_kernel<<<B_ * H_ * (S_ / 64), 128, 0, stream>>>(qT, kT, vT, ao);

  // 5) Output projection -> d_out (f32)
  gemm_bf16_f32_kernel<<<dim3(HID_ / 64, M / 128), 256, 0, stream>>>(
      ao, woT, out, M, HID_, H_ * D_);
}
